// NodeformerProcessor_28999619182946
// MI455X (gfx1250) — matrix operand order repaired and checked by reading, not compiled, not run on hardware
//
#include <hip/hip_runtime.h>
#include <hip/hip_bf16.h>
#include <math.h>

// ---------------- problem constants (from reference) ----------------
#define NNODES 50000
#define NEDGES 800000
#define CH     32          // channels
#define KSIZE  4           // spline kernel size per dim
#define NH     8           // heads
#define DHD    4           // head dim
#define NM     32          // random features
#define BN_BLOCKS 128
#define EPSBN  1e-5f
#define SLOPE  0.01f

// ---------------- WMMA types (gfx1250, wave32) ----------------
typedef __attribute__((ext_vector_type(16))) __bf16 v16bf;
typedef __attribute__((ext_vector_type(8)))  float  v8f;

union ABf { v16bf v; __bf16 e[16]; };
union CFr { v8f   v; float  e[8];  };

__device__ __forceinline__ float warp_sum32(float x) {
#pragma unroll
  for (int o = 16; o > 0; o >>= 1) x += __shfl_xor(x, o);
  return x;
}

// =====================================================================
// Degree kernels
// =====================================================================
__global__ void k_deg_accum(const int* __restrict__ ei, float* __restrict__ deg) {
  int e = blockIdx.x * blockDim.x + threadIdx.x;
  if (e < NEDGES) atomicAdd(&deg[ei[NEDGES + e]], 1.0f);
}

__global__ void k_deg_final(float* __restrict__ deg, float* __restrict__ rsd,
                            float* __restrict__ invdeg) {
  int n = blockIdx.x * blockDim.x + threadIdx.x;
  if (n < NNODES) {
    float d = deg[n];
    d = d < 1.0f ? 1.0f : d;
    deg[n]    = d;
    rsd[n]    = rsqrtf(d);
    invdeg[n] = 1.0f / d;     // one exact divide per node, reused everywhere
  }
}

__global__ void k_sig8(const float* __restrict__ brel, float* __restrict__ sigb) {
  int i = threadIdx.x;
  if (i < NH) sigb[i] = 1.0f / (1.0f + expf(-brel[i]));
}

// =====================================================================
// SplineConv edge kernel: wave per edge, lane = output channel d.
// Edge index is wave-uniform -> force to SGPR with readfirstlane so edge
// metadata / basis math / x[src,c] broadcasts become SALU + scalar loads;
// the w FMAs then use an SGPR multiplicand (no shuffles, no bpermute).
// w (256KB) stays resident in the 192MB L2.
// =====================================================================
__global__ void k_spline_edges(const float* __restrict__ x, const int* __restrict__ ei,
                               const float* __restrict__ attr, const float* __restrict__ w,
                               float* __restrict__ agg) {
  int lane  = threadIdx.x & 31;
  int gwave = (blockIdx.x * blockDim.x + threadIdx.x) >> 5;
  int nwave = (gridDim.x * blockDim.x) >> 5;
  for (int ev = gwave; ev < NEDGES; ev += nwave) {
    int e   = __builtin_amdgcn_readfirstlane(ev);        // wave-uniform -> SGPR
    int src = __builtin_amdgcn_readfirstlane(ei[e]);
    int dst = __builtin_amdgcn_readfirstlane(ei[NEDGES + e]);

    float p0 = attr[e * 3 + 0] * (KSIZE - 1);
    float p1 = attr[e * 3 + 1] * (KSIZE - 1);
    float p2 = attr[e * 3 + 2] * (KSIZE - 1);
    int f0 = (int)floorf(p0); f0 = f0 < 0 ? 0 : (f0 > KSIZE - 2 ? KSIZE - 2 : f0);
    int f1 = (int)floorf(p1); f1 = f1 < 0 ? 0 : (f1 > KSIZE - 2 ? KSIZE - 2 : f1);
    int f2 = (int)floorf(p2); f2 = f2 < 0 ? 0 : (f2 > KSIZE - 2 ? KSIZE - 2 : f2);
    float r0 = p0 - (float)f0, r1 = p1 - (float)f1, r2 = p2 - (float)f2;
    int base = f0 + 4 * f1 + 16 * f2;

    int   idx[8];
    float bas[8];
#pragma unroll
    for (int b = 0; b < 8; ++b) {
      int b0 = b & 1, b1 = (b >> 1) & 1, b2 = (b >> 2) & 1;
      idx[b] = base + b0 + 4 * b1 + 16 * b2;
      bas[b] = (b0 ? r0 : 1.0f - r0) * (b1 ? r1 : 1.0f - r1) * (b2 ? r2 : 1.0f - r2);
    }

    const float* xs = x + src * CH;                      // SGPR base
    float t[8] = {0.f, 0.f, 0.f, 0.f, 0.f, 0.f, 0.f, 0.f};
#pragma unroll 4
    for (int c = 0; c < CH; ++c) {
      float xc = xs[c];                                  // uniform -> scalar load
#pragma unroll
      for (int b = 0; b < 8; ++b)
        t[b] = fmaf(xc, w[idx[b] * (CH * CH) + c * CH + lane], t[b]);
    }
    float msg = 0.f;
#pragma unroll
    for (int b = 0; b < 8; ++b) msg = fmaf(bas[b], t[b], msg);

    atomicAdd(&agg[dst * CH + lane], msg);
  }
}

// =====================================================================
// Generic [N,32] x [32,32] GEMM via v_wmma_f32_16x16x32_bf16.
// One wave -> 16 output rows x 32 cols (two 16x16 tiles, one WMMA each).
// Compile-time epilogue: out = wmma(A@B) [+ aux*invdeg] [+ bias] [leaky].
// A fragment loaded as 4x float4 (global_load_b128).
// =====================================================================
template <bool AUX, bool BIAS, bool LEAKY>
__global__ void k_gemm32(const float* __restrict__ A, const float* __restrict__ B,
                         const float* __restrict__ bias, const float* __restrict__ aux,
                         const float* __restrict__ invdeg, float* __restrict__ out,
                         int nrows) {
  int tid  = blockIdx.x * blockDim.x + threadIdx.x;
  int wid  = tid >> 5;
  int lane = threadIdx.x & 31;
  int ntiles = (nrows + 15) >> 4;
  if (wid >= ntiles) return;                    // whole-wave exit: EXEC stays all-1s
  int m  = lane & 15;
  int hi = lane >> 4;
  int row = (wid << 4) + m;                     // nrows = 50000 is 16-divisible

  // A fragment: lane m holds row M=m; element j -> K = (j<8?j:j+8)+hi*8
  const float4* A4 = (const float4*)(A + row * CH);
  float4 q0 = A4[hi * 2 + 0];                   // K = hi*8   .. hi*8+3
  float4 q1 = A4[hi * 2 + 1];                   // K = hi*8+4 .. hi*8+7
  float4 q2 = A4[4 + hi * 2 + 0];               // K = 16+hi*8 ..
  float4 q3 = A4[4 + hi * 2 + 1];
  ABf a;
  a.e[0]  = (__bf16)q0.x; a.e[1]  = (__bf16)q0.y; a.e[2]  = (__bf16)q0.z; a.e[3]  = (__bf16)q0.w;
  a.e[4]  = (__bf16)q1.x; a.e[5]  = (__bf16)q1.y; a.e[6]  = (__bf16)q1.z; a.e[7]  = (__bf16)q1.w;
  a.e[8]  = (__bf16)q2.x; a.e[9]  = (__bf16)q2.y; a.e[10] = (__bf16)q2.z; a.e[11] = (__bf16)q2.w;
  a.e[12] = (__bf16)q3.x; a.e[13] = (__bf16)q3.y; a.e[14] = (__bf16)q3.z; a.e[15] = (__bf16)q3.w;

#pragma unroll
  for (int t = 0; t < 2; ++t) {
    int col = t * 16 + m;
    // B fragment: lane m holds col N=col; element j -> K = (j<8?j:j+8)+hi*8
    ABf b;
#pragma unroll
    for (int j = 0; j < 16; ++j) b.e[j] = (__bf16)B[((j < 8 ? j : j + 8) + hi * 8) * CH + col];

    CFr c;
#pragma unroll
    for (int i = 0; i < 8; ++i) c.e[i] = 0.f;
    c.v = __builtin_amdgcn_wmma_f32_16x16x32_bf16(
        false, a.v, false, b.v, (short)0, c.v, false, false);

    float bcol = BIAS ? bias[col] : 0.0f;       // hoisted out of element loop

    // C layout: element i -> row (i + hi*8), col m
#pragma unroll
    for (int i = 0; i < 8; ++i) {
      int r = (wid << 4) + i + hi * 8;
      float val = c.e[i];
      if (AUX) val = fmaf(aux[r * CH + col], invdeg[r], val);
      if (BIAS) val += bcol;
      if (LEAKY) val = val >= 0.f ? val : SLOPE * val;
      out[r * CH + col] = val;
    }
  }
}

// =====================================================================
// NodeFormer: KV summary reduction. Wave per node, lane = m (feature).
// phi_k computed in-register (never materialized); LDS partials -> global.
// =====================================================================
__global__ void k_nf_kv(const float* __restrict__ kbuf, const float* __restrict__ vbuf,
                        const float* __restrict__ wp, const float* __restrict__ gum,
                        float* __restrict__ kvg, float* __restrict__ ksg) {
  __shared__ float s_kv[NH * NM * DHD];   // 1024 floats
  __shared__ float s_ks[NH * NM];         // 256 floats
  for (int i = threadIdx.x; i < NH * NM * DHD; i += blockDim.x) s_kv[i] = 0.f;
  for (int i = threadIdx.x; i < NH * NM; i += blockDim.x) s_ks[i] = 0.f;
  __syncthreads();

  const float inv4  = 0.70710678118f;     // 1 / DH^0.25, DH=4
  const float invsm = 0.17677669529f;     // 1 / sqrt(M=32)
  int lane  = threadIdx.x & 31;
  int gwave = (blockIdx.x * blockDim.x + threadIdx.x) >> 5;
  int nwave = (gridDim.x * blockDim.x) >> 5;
  for (int n = gwave; n < NNODES; n += nwave) {
    float kb = kbuf[n * CH + lane] * inv4;
    float vv = vbuf[n * CH + lane];
#pragma unroll
    for (int h = 0; h < NH; ++h) {
      float x0 = __shfl(kb, h * 4 + 0), x1 = __shfl(kb, h * 4 + 1);
      float x2 = __shfl(kb, h * 4 + 2), x3 = __shfl(kb, h * 4 + 3);
      float sq = 0.5f * (x0 * x0 + x1 * x1 + x2 * x2 + x3 * x3);
      const float* wph = wp + h * (DHD * NM);
      float proj = x0 * wph[0 * NM + lane] + x1 * wph[1 * NM + lane] +
                   x2 * wph[2 * NM + lane] + x3 * wph[3 * NM + lane];
      float phi = expf(proj - sq) * invsm * expf(gum[n * NH + h]);   // tau = 1
      float v0 = __shfl(vv, h * 4 + 0), v1 = __shfl(vv, h * 4 + 1);
      float v2 = __shfl(vv, h * 4 + 2), v3 = __shfl(vv, h * 4 + 3);
      int b = (h * NM + lane) * DHD;
      atomicAdd(&s_kv[b + 0], phi * v0);
      atomicAdd(&s_kv[b + 1], phi * v1);
      atomicAdd(&s_kv[b + 2], phi * v2);
      atomicAdd(&s_kv[b + 3], phi * v3);
      atomicAdd(&s_ks[h * NM + lane], phi);
    }
  }
  __syncthreads();
  for (int i = threadIdx.x; i < NH * NM * DHD; i += blockDim.x) atomicAdd(&kvg[i], s_kv[i]);
  for (int i = threadIdx.x; i < NH * NM; i += blockDim.x) atomicAdd(&ksg[i], s_ks[i]);
}

// =====================================================================
// NodeFormer: attention numerator/denominator. Wave per node, lane = m.
// acc[n,h*4+d] = (phi_q . kv[h,:,d]) / (phi_q . ksum[h] + 1e-6)
// =====================================================================
__global__ void k_nf_attn(const float* __restrict__ qbuf, const float* __restrict__ wp,
                          const float* __restrict__ kvg, const float* __restrict__ ksg,
                          float* __restrict__ acc) {
  const float inv4  = 0.70710678118f;
  const float invsm = 0.17677669529f;
  int lane  = threadIdx.x & 31;
  int gwave = (blockIdx.x * blockDim.x + threadIdx.x) >> 5;
  int nwave = (gridDim.x * blockDim.x) >> 5;
  for (int n = gwave; n < NNODES; n += nwave) {
    float qv = qbuf[n * CH + lane] * inv4;
#pragma unroll
    for (int h = 0; h < NH; ++h) {
      float x0 = __shfl(qv, h * 4 + 0), x1 = __shfl(qv, h * 4 + 1);
      float x2 = __shfl(qv, h * 4 + 2), x3 = __shfl(qv, h * 4 + 3);
      float sq = 0.5f * (x0 * x0 + x1 * x1 + x2 * x2 + x3 * x3);
      const float* wph = wp + h * (DHD * NM);
      float proj = x0 * wph[0 * NM + lane] + x1 * wph[1 * NM + lane] +
                   x2 * wph[2 * NM + lane] + x3 * wph[3 * NM + lane];
      float phi = expf(proj - sq) * invsm;
      float den = warp_sum32(phi * ksg[h * NM + lane]) + 1e-6f;
      float rden = 1.0f / den;
      float at[DHD];
#pragma unroll
      for (int d = 0; d < DHD; ++d)
        at[d] = warp_sum32(phi * kvg[(h * NM + lane) * DHD + d]) * rden;
      if ((lane >> 2) == h) acc[n * CH + lane] = at[lane & 3];
    }
  }
}

// =====================================================================
// NodeFormer: edge relational bias. Wave per edge, lane = channel.
// Edge metadata scalarized; sigmoid(brel) precomputed (8 values).
// acc[dst,j] += sigb[h] * rsqrt(deg_s*deg_d) * v[src,j]
// =====================================================================
__global__ void k_nf_rel(const int* __restrict__ ei, const float* __restrict__ vbuf,
                         const float* __restrict__ sigb, const float* __restrict__ rsd,
                         float* __restrict__ acc) {
  int lane  = threadIdx.x & 31;
  int gwave = (blockIdx.x * blockDim.x + threadIdx.x) >> 5;
  int nwave = (gridDim.x * blockDim.x) >> 5;
  float sg = sigb[lane >> 2];
  for (int ev = gwave; ev < NEDGES; ev += nwave) {
    int e = __builtin_amdgcn_readfirstlane(ev);
    int s = __builtin_amdgcn_readfirstlane(ei[e]);
    int d = __builtin_amdgcn_readfirstlane(ei[NEDGES + e]);
    float nrm = rsd[s] * rsd[d];                     // scalar loads
    atomicAdd(&acc[d * CH + lane], sg * nrm * vbuf[s * CH + lane]);
  }
}

// =====================================================================
// BatchNorm (training-mode, biased var): deterministic two-level reduce.
// =====================================================================
__global__ void k_bn_partial(const float* __restrict__ z, float* __restrict__ psum,
                             float* __restrict__ psq) {
  __shared__ float ss[256], sq[256];
  int tid = threadIdx.x;
  int j = tid & 31, sub = tid >> 5;
  float ls = 0.f, lq = 0.f;
  for (int r = blockIdx.x * 8 + sub; r < NNODES; r += gridDim.x * 8) {
    float v = z[r * CH + j];
    ls += v; lq += v * v;
  }
  ss[tid] = ls; sq[tid] = lq;
  __syncthreads();
  for (int s = 128; s >= 32; s >>= 1) {
    if (tid < s) { ss[tid] += ss[tid + s]; sq[tid] += sq[tid + s]; }
    __syncthreads();
  }
  if (tid < 32) {
    psum[blockIdx.x * 32 + tid] = ss[tid];
    psq [blockIdx.x * 32 + tid] = sq[tid];
  }
}

__global__ void k_bn_final(const float* __restrict__ psum, const float* __restrict__ psq,
                           float* __restrict__ mean, float* __restrict__ rstd) {
  int j = threadIdx.x;
  if (j >= 32) return;
  float s = 0.f, q = 0.f;
  for (int b = 0; b < BN_BLOCKS; ++b) { s += psum[b * 32 + j]; q += psq[b * 32 + j]; }
  float mu  = s / (float)NNODES;
  float var = q / (float)NNODES - mu * mu;
  mean[j] = mu;
  rstd[j] = rsqrtf(var + EPSBN);
}

__global__ void k_bn_apply(const float* __restrict__ z, const float* __restrict__ mean,
                           const float* __restrict__ rstd, const float* __restrict__ gamma,
                           const float* __restrict__ beta, float* __restrict__ o) {
  int i = blockIdx.x * blockDim.x + threadIdx.x;
  if (i < NNODES * CH) {
    int j = i & 31;
    o[i] = (z[i] - mean[j]) * rstd[j] * gamma[j] + beta[j];
  }
}

// =====================================================================
// Host orchestration
// =====================================================================
extern "C" void kernel_launch(void* const* d_in, const int* in_sizes, int n_in,
                              void* d_out, int out_size, void* d_ws, size_t ws_size,
                              hipStream_t stream) {
  const float* x0    = (const float*)d_in[0];
  const int*   ei    = (const int*)  d_in[1];
  const float* attr  = (const float*)d_in[2];
  const float* wsp1  = (const float*)d_in[3];
  const float* root1 = (const float*)d_in[4];
  const float* bias1 = (const float*)d_in[5];
  const float* wsp2  = (const float*)d_in[6];
  const float* root2 = (const float*)d_in[7];
  const float* bias2 = (const float*)d_in[8];
  const float* bng   = (const float*)d_in[9];
  const float* bnb   = (const float*)d_in[10];
  const float* wqkv1 = (const float*)d_in[11];
  const float* wp1   = (const float*)d_in[12];
  const float* brel1 = (const float*)d_in[13];
  const float* wo1   = (const float*)d_in[14];
  const float* bo1   = (const float*)d_in[15];
  const float* g1    = (const float*)d_in[16];
  const float* wqkv2 = (const float*)d_in[17];
  const float* wp2   = (const float*)d_in[18];
  const float* brel2 = (const float*)d_in[19];
  const float* wo2   = (const float*)d_in[20];
  const float* bo2   = (const float*)d_in[21];
  const float* g2    = (const float*)d_in[22];
  float* out = (float*)d_out;

  // workspace layout (floats)
  float* ws    = (float*)d_ws;
  float* xA    = ws;                     // [N,32]
  float* xB    = xA   + NNODES * CH;     // [N,32]
  float* agg   = xB   + NNODES * CH;     // [N,32] spline agg / attn+rel acc
  float* qb    = agg  + NNODES * CH;     // [N,32]
  float* kb    = qb   + NNODES * CH;     // [N,32]
  float* vb    = kb   + NNODES * CH;     // [N,32]
  float* deg   = vb   + NNODES * CH;     // [N]
  float* rsd   = deg  + NNODES;          // [N]
  float* invd  = rsd  + NNODES;          // [N]
  float* kvb   = invd + NNODES;          // [H,M,DH] = 1024
  float* ksum  = kvb  + NH * NM * DHD;   // [H,M] = 256 (contiguous with kvb)
  float* sigb  = ksum + NH * NM;         // [8]
  float* mean  = sigb + 8;               // [32]
  float* rstd  = mean + 32;              // [32]
  float* psum  = rstd + 32;              // [BN_BLOCKS,32]
  float* psq   = psum + BN_BLOCKS * 32;  // [BN_BLOCKS,32]

  const int gemmBlocks = ((NNODES / 16) + 7) / 8;   // 8 waves/block, 1 wave/16 rows

  // ---- degrees (shared by spline layers + nodeformer rel) ----
  hipMemsetAsync(deg, 0, NNODES * sizeof(float), stream);
  k_deg_accum<<<(NEDGES + 255) / 256, 256, 0, stream>>>(ei, deg);
  k_deg_final<<<(NNODES + 255) / 256, 256, 0, stream>>>(deg, rsd, invd);

  auto run_bn = [&](int layer, const float* z, float* o) {
    k_bn_partial<<<BN_BLOCKS, 256, 0, stream>>>(z, psum, psq);
    k_bn_final<<<1, 32, 0, stream>>>(psum, psq, mean, rstd);
    k_bn_apply<<<(NNODES * CH + 255) / 256, 256, 0, stream>>>(
        z, mean, rstd, bng + layer * CH, bnb + layer * CH, o);
  };

  auto run_spline = [&](const float* xin, const float* wsp, const float* root,
                        const float* bias, float* zout) {
    hipMemsetAsync(agg, 0, NNODES * CH * sizeof(float), stream);
    k_spline_edges<<<2048, 256, 0, stream>>>(xin, ei, attr, wsp, agg);
    // z = leaky( wmma(x@root) + agg*invdeg + bias )
    k_gemm32<true, true, true><<<gemmBlocks, 256, 0, stream>>>(
        xin, root, bias, agg, invd, zout, NNODES);
  };

  auto run_nf = [&](const float* xin, const float* wqkv, const float* wp,
                    const float* brel, const float* wo, const float* bo,
                    const float* gum, float* zout, bool leaky) {
    k_gemm32<false, false, false><<<gemmBlocks, 256, 0, stream>>>(
        xin, wqkv + 0,    nullptr, nullptr, nullptr, qb, NNODES);
    k_gemm32<false, false, false><<<gemmBlocks, 256, 0, stream>>>(
        xin, wqkv + 1024, nullptr, nullptr, nullptr, kb, NNODES);
    k_gemm32<false, false, false><<<gemmBlocks, 256, 0, stream>>>(
        xin, wqkv + 2048, nullptr, nullptr, nullptr, vb, NNODES);
    hipMemsetAsync(kvb, 0, (NH * NM * DHD + NH * NM) * sizeof(float), stream);
    k_sig8<<<1, 32, 0, stream>>>(brel, sigb);
    k_nf_kv  <<<512, 256, 0, stream>>>(kb, vb, wp, gum, kvb, ksum);
    k_nf_attn<<<512, 256, 0, stream>>>(qb, wp, kvb, ksum, agg);
    k_nf_rel <<<2048, 256, 0, stream>>>(ei, vb, sigb, rsd, agg);
    if (leaky)
      k_gemm32<false, true, true><<<gemmBlocks, 256, 0, stream>>>(
          agg, wo, bo, nullptr, nullptr, zout, NNODES);
    else
      k_gemm32<false, true, false><<<gemmBlocks, 256, 0, stream>>>(
          agg, wo, bo, nullptr, nullptr, zout, NNODES);
  };

  // ---- layer 1: spline + leaky + BN0 ----
  run_spline(x0, wsp1, root1, bias1, xB);
  run_bn(0, xB, xA);
  // ---- layer 2: spline + leaky + BN1 ----
  run_spline(xA, wsp2, root2, bias2, xB);
  run_bn(1, xB, xA);
  // ---- layer 3: nodeformer + leaky + BN2 ----
  run_nf(xA, wqkv1, wp1, brel1, wo1, bo1, g1, xB, true);
  run_bn(2, xB, xA);
  // ---- layer 4: nodeformer + BN3 -> out ----
  run_nf(xA, wqkv2, wp2, brel2, wo2, bo2, g2, xB, false);
  run_bn(3, xB, out);
}
